// CustomMultiInputLSTM_24386824307334
// MI455X (gfx1250) — matrix operand order repaired and checked by reading, not compiled, not run on hardware
//
#include <hip/hip_runtime.h>
#include <hip/hip_bf16.h>

typedef __bf16 bf16_t;
typedef __attribute__((ext_vector_type(16))) __bf16 v16bf;
typedef __attribute__((ext_vector_type(8)))  __bf16 v8bf;
typedef __attribute__((ext_vector_type(8)))  float  v8f;

#define B_SZ 256
#define T_SZ 512
#define D_SZ 128
#define H_SZ 512
#define BM   16               // batch rows per workgroup
#define NT   (H_SZ/16)        // 32 column tiles of 16
#define KT_H (H_SZ/32)        // 16 k-tiles for recurrent GEMMs (K=512)
#define KT_D (D_SZ/32)        // 4 k-tiles for input GEMMs (K=128)
#define NTW  2                // n-tiles per wave (16 waves * 2 = 32)
#define LSTR 520              // padded LDS row stride (bf16 elems), 16B aligned

// gate order: 0=i 1=i_p 2=i_n 3=f 4=c 5=c_p 6=c_n 7=o ; U slot 8 = W_a
#define WPK_ELEMS ((size_t)8*KT_D*NT*512)
#define UPK_ELEMS ((size_t)9*KT_H*NT*512)

// ---------------------------------------------------------------- helpers
__device__ inline v8f splat8(float v) {
  v8f r;
#pragma unroll
  for (int i = 0; i < 8; ++i) r[i] = v;
  return r;
}

__device__ inline float fast_sig(float x) {
  return 1.f / (1.f + __expf(-x));
}
__device__ inline float fast_tanh(float x) {
  x = fminf(fmaxf(x, -15.f), 15.f);
  float e = __expf(2.f * x);
  return (e - 1.f) / (e + 1.f);
}

// B fragment: 32x16 bf16 tile, packed lane-contiguous: 16 bf16 (32B) per lane.
__device__ inline v16bf load_b_frag(const bf16_t* pk, int kt, int nt, int lane) {
  return *(const v16bf*)(pk + ((size_t)(kt * NT + nt) * 32 + lane) * 16);
}

// A fragment from LDS row-major bf16 [BM][LSTR]:
// lanes 0-15: row=lane,    K = {0..7, 16..23}  of the 32-wide k-tile
// lanes 16-31: row=lane-16, K = {8..15, 24..31}
__device__ inline v16bf load_a_lds(const bf16_t* base, int kt, int lrow, int hi) {
  const bf16_t* p = base + lrow * LSTR + kt * 32 + (hi ? 8 : 0);
  v8bf lo = *(const v8bf*)p;
  v8bf hg = *(const v8bf*)(p + 16);
  v16bf r;
#pragma unroll
  for (int i = 0; i < 8; ++i) { r[i] = lo[i]; r[8 + i] = hg[i]; }
  return r;
}

// A fragment from global f32 stream (converted to bf16 on the fly).
__device__ inline v16bf load_a_gf(const float* p, int hi) {
  const float* q = p + (hi ? 8 : 0);
  v16bf r;
#pragma unroll
  for (int i = 0; i < 8; ++i) {
    r[i]     = (bf16_t)q[i];
    r[8 + i] = (bf16_t)q[16 + i];
  }
  return r;
}

// ---------------------------------------------------------------- weight repack
struct PackArgs {
  const float* w[8];  // (D,H) row-major
  const float* u[9];  // (H,H) row-major, slot 8 = W_a
};

__global__ void pack_weights(PackArgs args, bf16_t* __restrict__ wpk,
                             bf16_t* __restrict__ upk) {
  const int tile = blockIdx.x;
  const int tid  = threadIdx.x;          // 0..511
  const int lane = tid >> 4;             // 0..31
  const int idx  = tid & 15;             // 2*v + h (v16bf element index)
  const int v = idx >> 1, h = idx & 1;
  const int n_local = lane & 15;
  const int k_local = ((idx >> 3) ? 16 : 0) + ((lane >> 4) ? 8 : 0) + (idx & 7);
  const int WTILES = 8 * KT_D * NT;
  if (tile < WTILES) {
    int g  = tile / (KT_D * NT);
    int r  = tile % (KT_D * NT);
    int kt = r / NT, nt = r % NT;
    int k = kt * 32 + k_local, n = nt * 16 + n_local;
    wpk[(size_t)tile * 512 + lane * 16 + idx] =
        (bf16_t)args.w[g][(size_t)k * H_SZ + n];
  } else {
    int t2 = tile - WTILES;
    int m  = t2 / (KT_H * NT);
    int r  = t2 % (KT_H * NT);
    int kt = r / NT, nt = r % NT;
    int k = kt * 32 + k_local, n = nt * 16 + n_local;
    upk[(size_t)t2 * 512 + lane * 16 + idx] =
        (bf16_t)args.u[m][(size_t)k * H_SZ + n];
  }
}

// ---------------------------------------------------------------- recurrent kernel
// 16 workgroups (one per 16-row batch block), 512 threads = 16 wave32s
// (4 waves/SIMD -> full 256-VGPR budget per wave). Each wave owns two
// 16-column tiles of H. The recurrence is independent per batch row, so only
// workgroup barriers are needed (l_sm / h_sm publication).
__global__ __launch_bounds__(512, 1)
void lstm_kernel(const float* __restrict__ Yg, const float* __restrict__ Pg,
                 const float* __restrict__ Ng,
                 const float* __restrict__ b_i,  const float* __restrict__ b_ip,
                 const float* __restrict__ b_in, const float* __restrict__ b_f,
                 const float* __restrict__ b_c,  const float* __restrict__ b_cp,
                 const float* __restrict__ b_cn, const float* __restrict__ b_o,
                 const float* __restrict__ b_a,
                 const bf16_t* __restrict__ wpk, const bf16_t* __restrict__ upk,
                 float* __restrict__ hlast, float* __restrict__ hseq) {
  __shared__ bf16_t h_sm[BM * LSTR];
  __shared__ bf16_t l_sm[3][BM * LSTR];

  const int tid  = threadIdx.x;
  const int lane = tid & 31;
  const int wave = tid >> 5;    // 0..15
  const int nt0  = wave * NTW;  // first owned n-tile
  const int lrow = lane & 15;   // row (A/C) or column (B / C-N) inside tile
  const int hi   = lane >> 4;   // half select
  const int b0   = blockIdx.x * BM;
  const int Mrow = (hi ? 8 : 0);  // C/D vgpr i -> M = i + Mrow

  for (int i = tid; i < BM * LSTR; i += 512) h_sm[i] = (bf16_t)0.f;
  __syncthreads();

  v8f c_acc[NTW];
#pragma unroll
  for (int n = 0; n < NTW; ++n) c_acc[n] = splat8(0.f);

  for (int t = 0; t < T_SZ; ++t) {
    const size_t xoff = ((size_t)(b0 + lrow) * T_SZ + t) * D_SZ;

    // Two gate pre-activations sharing the A fragments (x-GEMM K=128 from
    // stream Xg, then h-GEMM K=512 from LDS h).
    auto gate_pre2 = [&](const float* Xg, int gA, int gB, const float* biasA,
                         const float* biasB, v8f* accA, v8f* accB) {
      const bf16_t* wA = wpk + (size_t)gA * KT_D * NT * 512;
      const bf16_t* wB = wpk + (size_t)gB * KT_D * NT * 512;
      const bf16_t* uA = upk + (size_t)gA * KT_H * NT * 512;
      const bf16_t* uB = upk + (size_t)gB * KT_H * NT * 512;
#pragma unroll
      for (int n = 0; n < NTW; ++n) {
        accA[n] = splat8(biasA[(nt0 + n) * 16 + lrow]);
        accB[n] = splat8(biasB[(nt0 + n) * 16 + lrow]);
      }
#pragma clang loop unroll_count(2)
      for (int kt = 0; kt < KT_D; ++kt) {
        v16bf a = load_a_gf(Xg + xoff + kt * 32, hi);
#pragma unroll
        for (int n = 0; n < NTW; ++n) {
          accA[n] = __builtin_amdgcn_wmma_f32_16x16x32_bf16(
              false, a, false, load_b_frag(wA, kt, nt0 + n, lane), (short)0,
              accA[n], false, false);
          accB[n] = __builtin_amdgcn_wmma_f32_16x16x32_bf16(
              false, a, false, load_b_frag(wB, kt, nt0 + n, lane), (short)0,
              accB[n], false, false);
        }
      }
#pragma clang loop unroll_count(4)
      for (int kt = 0; kt < KT_H; ++kt) {
        v16bf ah = load_a_lds(h_sm, kt, lrow, hi);
#pragma unroll
        for (int n = 0; n < NTW; ++n) {
          accA[n] = __builtin_amdgcn_wmma_f32_16x16x32_bf16(
              false, ah, false, load_b_frag(uA, kt, nt0 + n, lane), (short)0,
              accA[n], false, false);
          accB[n] = __builtin_amdgcn_wmma_f32_16x16x32_bf16(
              false, ah, false, load_b_frag(uB, kt, nt0 + n, lane), (short)0,
              accB[n], false, false);
        }
      }
    };

    // l_s = tanh(c_pre) * sigmoid(i_pre), published to LDS only (no register
    // copies survive this scope).
    auto make_l = [&](const float* Xg, int gI, int gC, const float* bI,
                      const float* bC, int s) {
      v8f pi[NTW], pc[NTW];
      gate_pre2(Xg, gI, gC, bI, bC, pi, pc);
#pragma unroll
      for (int n = 0; n < NTW; ++n) {
        const int ncol = (nt0 + n) * 16 + lrow;
#pragma unroll
        for (int i = 0; i < 8; ++i) {
          float lv = fast_tanh(pc[n][i]) * fast_sig(pi[n][i]);
          l_sm[s][(i + Mrow) * LSTR + ncol] = (bf16_t)lv;
        }
      }
    };

    make_l(Yg, 0, 4, b_i,  b_c,  0);   // i , c   (stream Y)
    make_l(Pg, 1, 5, b_ip, b_cp, 1);   // i_p,c_p (stream P)
    make_l(Ng, 2, 6, b_in, b_cn, 2);   // i_n,c_n (stream N)

    v8f ft[NTW], ot[NTW];
    {
      v8f pf[NTW], po[NTW];
      gate_pre2(Yg, 3, 7, b_f, b_o, pf, po);   // f , o (stream Y)
#pragma unroll
      for (int n = 0; n < NTW; ++n)
#pragma unroll
        for (int i = 0; i < 8; ++i) {
          ft[n][i] = fast_sig(pf[n][i]);
          ot[n][i] = fast_sig(po[n][i]);
        }
    }

    __syncthreads();  // l_sm ready; all h_sm reads of this step done

    // u_s = tanh((l_s @ W_a) * c + b_a), consumed by ONLINE softmax:
    // keep running (max m, denom sum, weighted numerator Lc) -- no u array.
    const bf16_t* wa = upk + (size_t)8 * KT_H * NT * 512;
    v8f mx[NTW], sum[NTW], Lc[NTW];
#pragma unroll
    for (int s = 0; s < 3; ++s) {
      v8f acc[NTW];
#pragma unroll
      for (int n = 0; n < NTW; ++n) acc[n] = splat8(0.f);
#pragma clang loop unroll_count(4)
      for (int kt = 0; kt < KT_H; ++kt) {
        v16bf al = load_a_lds(l_sm[s], kt, lrow, hi);
#pragma unroll
        for (int n = 0; n < NTW; ++n)
          acc[n] = __builtin_amdgcn_wmma_f32_16x16x32_bf16(
              false, al, false, load_b_frag(wa, kt, nt0 + n, lane), (short)0,
              acc[n], false, false);
      }
#pragma unroll
      for (int n = 0; n < NTW; ++n) {
        const int ncol = (nt0 + n) * 16 + lrow;
        const float bav = b_a[ncol];
#pragma unroll
        for (int i = 0; i < 8; ++i) {
          float u = fast_tanh(acc[n][i] * c_acc[n][i] + bav);
          float lv = (float)l_sm[s][(i + Mrow) * LSTR + ncol];
          if (s == 0) {
            mx[n][i] = u;
            sum[n][i] = 1.f;
            Lc[n][i] = lv;
          } else {
            float mn = fmaxf(mx[n][i], u);
            float sc = __expf(mx[n][i] - mn);
            float e  = __expf(u - mn);
            sum[n][i] = sum[n][i] * sc + e;
            Lc[n][i]  = Lc[n][i] * sc + e * lv;
            mx[n][i]  = mn;
          }
        }
      }
    }

    // cell / hidden update, store outputs
#pragma unroll
    for (int n = 0; n < NTW; ++n) {
      const int ncol = (nt0 + n) * 16 + lrow;
#pragma unroll
      for (int i = 0; i < 8; ++i) {
        const int M = i + Mrow;
        float cn = ft[n][i] * c_acc[n][i] + Lc[n][i] / sum[n][i];
        c_acc[n][i] = cn;
        float hn = ot[n][i] * fast_tanh(cn);
        hseq[((size_t)(b0 + M) * T_SZ + t) * H_SZ + ncol] = hn;
        h_sm[M * LSTR + ncol] = (bf16_t)hn;
        if (t == T_SZ - 1) hlast[(size_t)(b0 + M) * H_SZ + ncol] = hn;
      }
    }
    __syncthreads();  // h_sm published for next step
  }
}

// ---------------------------------------------------------------- launch
extern "C" void kernel_launch(void* const* d_in, const int* in_sizes, int n_in,
                              void* d_out, int out_size, void* d_ws,
                              size_t ws_size, hipStream_t stream) {
  (void)in_sizes; (void)n_in; (void)out_size; (void)ws_size;

  PackArgs pa;
  const int wi[8] = {0, 3, 6, 9, 12, 15, 18, 21};            // W_i..W_o
  const int ui[9] = {1, 4, 7, 10, 13, 16, 19, 22, 24};       // U_i..U_o, W_a
  for (int g = 0; g < 8; ++g) pa.w[g] = (const float*)d_in[wi[g]];
  for (int m = 0; m < 9; ++m) pa.u[m] = (const float*)d_in[ui[m]];

  bf16_t* wpk = (bf16_t*)d_ws;
  bf16_t* upk = wpk + WPK_ELEMS;

  const int packBlocks = 8 * KT_D * NT + 9 * KT_H * NT;
  pack_weights<<<packBlocks, 512, 0, stream>>>(pa, wpk, upk);

  float* out   = (float*)d_out;
  float* hlast = out;                              // [B, H]
  float* hseq  = out + (size_t)B_SZ * H_SZ;        // [B, T, H]

  lstm_kernel<<<B_SZ / BM, 512, 0, stream>>>(
      (const float*)d_in[26], (const float*)d_in[27], (const float*)d_in[28],
      (const float*)d_in[2],  (const float*)d_in[5],  (const float*)d_in[8],
      (const float*)d_in[11], (const float*)d_in[14], (const float*)d_in[17],
      (const float*)d_in[20], (const float*)d_in[23], (const float*)d_in[25],
      wpk, upk, hlast, hseq);
}
